// ImputedMultiTimeAttentionV2_45148696215775
// MI455X (gfx1250) — matrix operand order repaired and checked by reading, not compiled, not run on hardware
//
#include <hip/hip_runtime.h>
#include <hip/hip_bf16.h>

typedef __attribute__((ext_vector_type(16))) _Float16 v16h;
typedef __attribute__((ext_vector_type(8)))  float    v8f;
typedef __attribute__((ext_vector_type(4)))  unsigned int u32x4;
typedef __attribute__((ext_vector_type(4)))  float    f32x4;

#define B_  8
#define LQ_ 256
#define LK_ 512
#define D_  128
#define F_  32
#define H_  8
#define DK_ 16
#define QT  64

// ---- LDS layout (bytes) ----
// Qf16 : QT*32 halves                        @      0 (4096)
// Kf16 : LK*32 halves                        @   4096 (32768)
// Ut   : 64*LK halves (mask*V | mask)        @  36864 (65536)
// Wf   : QT*LK halves (UNION: f16 weights)   @ 102400 (65536)
//    WqT16 [DK][D] (4096) | WkT16 [DK][D] (4096) | WvT16 [F][F] (2048)
// red  : QT*2 f32                            @ 167936 (512)
#define SMEM_BYTES 168448

__device__ __forceinline__ v8f wmma_f16(v16h a, v16h b, v8f c) {
    return __builtin_amdgcn_wmma_f32_16x16x32_f16(false, a, false, b, (short)0, c,
                                                  false, false);
}

// A-fragment (16x32 f16) built from an LDS f16 row-major matrix:
// halves 0-7 at p0, halves 8-15 at p1.
__device__ __forceinline__ v16h frag2(const _Float16* p0, const _Float16* p1) {
    union { u32x4 u[2]; v16h v; } t;
    t.u[0] = *(const u32x4*)p0;
    t.u[1] = *(const u32x4*)p1;
    return t.v;
}

// A-fragment (16x32 f16) built directly from a global row-major f32 row.
// Lane layout: halves 0-7 = K kbase+lhalf*8+0..7, halves 8-15 = K kbase+16+lhalf*8+0..7.
__device__ __forceinline__ v16h fragA_g32(const float* row, int kbase, int lhalf) {
    const f32x4* p0 = (const f32x4*)(row + kbase + lhalf * 8);
    const f32x4* p1 = (const f32x4*)(row + kbase + 16 + lhalf * 8);
    f32x4 a0 = p0[0], a1 = p0[1];
    f32x4 b0 = p1[0], b1 = p1[1];
    union { _Float16 h[16]; v16h v; } t;
    t.h[0]  = (_Float16)a0.x; t.h[1]  = (_Float16)a0.y;
    t.h[2]  = (_Float16)a0.z; t.h[3]  = (_Float16)a0.w;
    t.h[4]  = (_Float16)a1.x; t.h[5]  = (_Float16)a1.y;
    t.h[6]  = (_Float16)a1.z; t.h[7]  = (_Float16)a1.w;
    t.h[8]  = (_Float16)b0.x; t.h[9]  = (_Float16)b0.y;
    t.h[10] = (_Float16)b0.z; t.h[11] = (_Float16)b0.w;
    t.h[12] = (_Float16)b1.x; t.h[13] = (_Float16)b1.y;
    t.h[14] = (_Float16)b1.z; t.h[15] = (_Float16)b1.w;
    return t.v;
}

extern "C" __global__ __launch_bounds__(256)
void attn_heads_kernel(const float* __restrict__ query,
                       const float* __restrict__ key,
                       const float* __restrict__ value,
                       const float* __restrict__ mask,
                       const float* __restrict__ Wq,
                       const float* __restrict__ Wk,
                       const float* __restrict__ Wv,
                       float* __restrict__ lat)
{
    extern __shared__ char smem[];
    _Float16* Qf    = (_Float16*)(smem);
    _Float16* Kf    = (_Float16*)(smem + 4096);
    _Float16* Ut    = (_Float16*)(smem + 36864);
    _Float16* Wf    = (_Float16*)(smem + 102400);
    _Float16* WqT16 = (_Float16*)(smem + 102400);          // [DK][D]
    _Float16* WkT16 = (_Float16*)(smem + 102400 + 4096);   // [DK][D]
    _Float16* WvT16 = (_Float16*)(smem + 102400 + 8192);   // [F][F] (f-major)
    float*    red   = (float*)(smem + 167936);             // [QT][2]

    const int tid   = threadIdx.x;
    const int qt    = blockIdx.x;   // q-tile 0..3
    const int j     = blockIdx.y;   // head
    const int b     = blockIdx.z;   // batch
    const int wv    = tid >> 5;
    const int lane  = tid & 31;
    const int lrow  = lane & 15;
    const int lhalf = lane >> 4;

    const v8f vzero = {0.f, 0.f, 0.f, 0.f, 0.f, 0.f, 0.f, 0.f};

    // ---------- Phase A: stage f16 weights (transposed) + zero padded operands ----------
    for (int idx = tid; idx < DK_ * D_; idx += 256) {
        int dk = idx >> 7, d = idx & 127;
        WqT16[idx] = (_Float16)Wq[((size_t)j * D_ + d) * DK_ + dk];
        WkT16[idx] = (_Float16)Wk[((size_t)j * D_ + d) * DK_ + dk];
    }
    for (int idx = tid; idx < F_ * F_; idx += 256) {
        int f = idx >> 5, f2 = idx & 31;
        WvT16[idx] = (_Float16)Wv[((size_t)j * F_ + f2) * F_ + f];
    }
    {   // zero Qf + Kf (covers the DK 16->32 zero padding)
        int* zp = (int*)smem;
        for (int idx = tid; idx < 9216; idx += 256) zp[idx] = 0;
    }
    __syncthreads();

    // ---------- Phase B: WMMA projections ----------
    // Q projection: waves 0-3, one 16-row tile each; Kdim = D = 4 chunks of 32
    if (wv < 4) {
        const int qb = wv;
        const float* arow = query + ((size_t)(b * LQ_ + qt * QT + qb * 16 + lrow)) * D_;
        v8f acc = vzero;
        #pragma unroll
        for (int c = 0; c < 4; ++c) {
            v16h afr = fragA_g32(arow, c * 32, lhalf);
            const _Float16* bp = WqT16 + lrow * D_ + c * 32 + lhalf * 8;
            v16h bfr = frag2(bp, bp + 16);
            acc = wmma_f16(afr, bfr, acc);
        }
        #pragma unroll
        for (int r = 0; r < 8; ++r)
            Qf[(qb * 16 + lhalf * 8 + r) * 32 + lrow] = (_Float16)acc[r];
    }
    // K projection: 4 tiles (64 k-rows) per wave
    #pragma unroll
    for (int t4 = 0; t4 < 4; ++t4) {
        const int kt = wv * 4 + t4;
        const float* arow = key + ((size_t)(b * LK_ + kt * 16 + lrow)) * D_;
        v8f acc = vzero;
        #pragma unroll
        for (int c = 0; c < 4; ++c) {
            v16h afr = fragA_g32(arow, c * 32, lhalf);
            const _Float16* bp = WkT16 + lrow * D_ + c * 32 + lhalf * 8;
            v16h bfr = frag2(bp, bp + 16);
            acc = wmma_f16(afr, bfr, acc);
        }
        #pragma unroll
        for (int r = 0; r < 8; ++r)
            Kf[(kt * 16 + lhalf * 8 + r) * 32 + lrow] = (_Float16)acc[r];
    }
    // V projection (Kdim = F = one chunk of 32) + mask, stored transposed in Ut
    #pragma unroll
    for (int t4 = 0; t4 < 4; ++t4) {
        const int kt = wv * 4 + t4;
        const float* arow = value + ((size_t)(b * LK_ + kt * 16 + lrow)) * F_;
        v16h afr = fragA_g32(arow, 0, lhalf);
        #pragma unroll
        for (int fb = 0; fb < 2; ++fb) {
            const _Float16* bp = WvT16 + (fb * 16 + lrow) * F_ + lhalf * 8;
            v16h bfr = frag2(bp, bp + 16);
            v8f acc = wmma_f16(afr, bfr, vzero);
            #pragma unroll
            for (int r = 0; r < 8; ++r) {
                int k = kt * 16 + lhalf * 8 + r;
                int f = fb * 16 + lrow;
                float mk = mask[((size_t)(b * LK_ + k)) * F_ + f];
                Ut[f * LK_ + k]        = (_Float16)(mk * acc[r]);
                Ut[(F_ + f) * LK_ + k] = (_Float16)mk;
            }
        }
    }
    __syncthreads();

    // ---------- Phase C: S = Q*K^T (WMMA), row max, W = exp(S - max) ----------
    {
        const int qb = wv >> 1;   // q block 0..3 (16 rows each)
        const int kh = wv & 1;    // k half: cols kh*256 .. +255

        const _Float16* ap = Qf + (qb * 16 + lrow) * 32;
        v16h afr = frag2(ap + lhalf * 8, ap + 16 + lhalf * 8);

        v8f acc[16];
        #pragma unroll
        for (int kb2 = 0; kb2 < 16; ++kb2) {
            const int kb = kh * 16 + kb2;
            const _Float16* bp = Kf + (kb * 16 + lrow) * 32 + lhalf * 8;
            v16h bfr = frag2(bp, bp + 16);
            acc[kb2] = wmma_f16(afr, bfr, vzero);
        }

        // per-row max over this wave's 256 columns
        float m[8];
        #pragma unroll
        for (int r = 0; r < 8; ++r) {
            float mm = acc[0][r];
            #pragma unroll
            for (int kb2 = 1; kb2 < 16; ++kb2) mm = fmaxf(mm, acc[kb2][r]);
            #pragma unroll
            for (int off = 1; off < 16; off <<= 1)
                mm = fmaxf(mm, __shfl_xor(mm, off, 32));
            m[r] = mm;   // row = qb*16 + lhalf*8 + r
        }
        if (lrow == 0) {
            #pragma unroll
            for (int r = 0; r < 8; ++r)
                red[(qb * 16 + lhalf * 8 + r) * 2 + kh] = m[r];
        }
        __syncthreads();

        float rm[8];
        #pragma unroll
        for (int r = 0; r < 8; ++r) {
            int row = qb * 16 + lhalf * 8 + r;
            rm[r] = fmaxf(red[row * 2], red[row * 2 + 1]);
        }
        #pragma unroll
        for (int kb2 = 0; kb2 < 16; ++kb2) {
            const int col = kh * 256 + kb2 * 16 + lrow;
            #pragma unroll
            for (int r = 0; r < 8; ++r) {
                int row = qb * 16 + lhalf * 8 + r;
                Wf[row * LK_ + col] = (_Float16)__expf(acc[kb2][r] - rm[r]);
            }
        }
    }
    __syncthreads();

    // ---------- Phase F: [numer|denom] = W @ [mask*V | mask] (WMMA), divide ----------
    {
        const int qb = wv >> 1;   // 0..3
        const int fp = wv & 1;    // feature block pair 0..1
        v8f accN = vzero;
        v8f accD = vzero;
        #pragma unroll 4
        for (int kc = 0; kc < 16; ++kc) {
            const _Float16* ap = Wf + (qb * 16 + lrow) * LK_ + kc * 32 + lhalf * 8;
            v16h afr = frag2(ap, ap + 16);
            const _Float16* bpn = Ut + (fp * 16 + lrow) * LK_ + kc * 32 + lhalf * 8;
            v16h bn = frag2(bpn, bpn + 16);
            const _Float16* bpd = bpn + 32 * LK_;
            v16h bd = frag2(bpd, bpd + 16);
            accN = wmma_f16(afr, bn, accN);
            accD = wmma_f16(afr, bd, accD);
        }
        #pragma unroll
        for (int r = 0; r < 8; ++r) {
            int q = qt * QT + qb * 16 + lhalf * 8 + r;
            int f = fp * 16 + lrow;
            float o = accN[r] / accD[r];
            lat[(((size_t)(b * LQ_ + q)) * F_ + f) * H_ + j] = o;
        }
    }
}

extern "C" __global__ __launch_bounds__(256)
void combine_kernel(const float* __restrict__ lat, const float* __restrict__ Wc,
                    const float* __restrict__ bc, float* __restrict__ out)
{
    int idx = blockIdx.x * 256 + threadIdx.x;   // < B*LQ*F
    int f = idx & (F_ - 1);
    float s = bc[f];
    const float* lp = lat + (size_t)idx * H_;
    const float* wp = Wc + f * H_;
    #pragma unroll
    for (int jj = 0; jj < H_; ++jj) s += lp[jj] * wp[jj];
    out[idx] = tanhf(s);
}

extern "C" void kernel_launch(void* const* d_in, const int* in_sizes, int n_in,
                              void* d_out, int out_size, void* d_ws, size_t ws_size,
                              hipStream_t stream) {
    const float* query = (const float*)d_in[0];
    const float* key   = (const float*)d_in[1];
    const float* value = (const float*)d_in[2];
    const float* mask  = (const float*)d_in[3];
    const float* Wq    = (const float*)d_in[4];
    const float* Wk    = (const float*)d_in[5];
    const float* Wv    = (const float*)d_in[6];
    const float* Wc    = (const float*)d_in[7];
    const float* bc    = (const float*)d_in[8];

    float* lat = (float*)d_ws;   // B*LQ*F*H floats = 2 MB

    dim3 grid(LQ_ / QT, H_, B_), blk(256, 1, 1);
    hipLaunchKernelGGL(attn_heads_kernel, grid, blk, SMEM_BYTES, stream,
                       query, key, value, mask, Wq, Wk, Wv, lat);

    combine_kernel<<<(B_ * LQ_ * F_) / 256, 256, 0, stream>>>(lat, Wc, bc, (float*)d_out);
}
